// AttnBlock_45835890983031
// MI455X (gfx1250) — matrix operand order repaired and checked by reading, not compiled, not run on hardware
//
#include <hip/hip_runtime.h>

// ---------------------------------------------------------------------------
// AttnBlock for MI455X (gfx1250): bf16 WMMA pipeline, f32 accumulate.
//   q = x Wq^T + bq ; k = ... ; v = ... ; S = q k^T ; P = softmax(S)
//   h = P v ; out = x + h Wo^T + bo
// Wave tile 64x64 (4x4 v_wmma_f32_16x16x32_bf16), block tile 256x128 (8 waves).
// K loop unrolled x2 so fragment buffers alternate (kills WMMA WAR v_nops).
// ---------------------------------------------------------------------------

typedef __attribute__((ext_vector_type(16))) __bf16 v16bf;
typedef __attribute__((ext_vector_type(8)))  __bf16 v8bf;
typedef __attribute__((ext_vector_type(8)))  float  v8f;

#define NTOK 4096
#define CDIM 1024

// ------------------------------ fp32 -> bf16 -------------------------------
__global__ __launch_bounds__(256) void cvt_f32_bf16(const float* __restrict__ src,
                                                    __bf16* __restrict__ dst, int n) {
  int i = blockIdx.x * blockDim.x + threadIdx.x;
  int stride = gridDim.x * blockDim.x;
  for (; i < n; i += stride) dst[i] = (__bf16)src[i];
}

// --------------------------- WMMA fragment loads ---------------------------
// A-matrix 16x32 bf16 (ISA 7.12.2): lane<16 -> row m0+lane, K {0..7,16..23};
// lane>=16 -> row m0+lane-16, K {8..15,24..31}. Two 16B loads per lane.
__device__ __forceinline__ v16bf load_fragA(const __bf16* __restrict__ p, int ld,
                                            int r0, int k0, int lane) {
  int r  = r0 + (lane & 15);
  int kh = k0 + ((lane >> 4) << 3);
  const __bf16* base = p + (size_t)r * ld + kh;
  v8bf lo = *(const v8bf*)(base);
  v8bf hi = *(const v8bf*)(base + 16);
  return __builtin_shufflevector(lo, hi, 0,1,2,3,4,5,6,7,8,9,10,11,12,13,14,15);
}

// B-matrix 32x16 bf16: lane<16 -> column n0+lane holds K {0..7,16..23};
// lane>=16 -> K {8..15,24..31} (two 16B loads per lane). Source is W[N,K] row-major.
__device__ __forceinline__ v16bf load_fragB(const __bf16* __restrict__ p, int ld,
                                            int r0, int k0, int lane) {
  int r  = r0 + (lane & 15);
  int kh = k0 + ((lane >> 4) << 3);
  const __bf16* base = p + (size_t)r * ld + kh;
  v8bf lo = *(const v8bf*)(base);
  v8bf hi = *(const v8bf*)(base + 16);
  return __builtin_shufflevector(lo, hi, 0,1,2,3,4,5,6,7,8,9,10,11,12,13,14,15);
}

// ------------------------------- NT GEMM -----------------------------------
// D[M,N] = A[M,K](bf16,row-major) * B[N,K]^T(bf16,row-major) (+bias[n]) (+resid)
// OUTM: 0 = fp32 out, 1 = bf16 out, 2 = bf16 out transposed (D[n*ldd+m]).
// Block = 8 waves (4 m x 2 n); wave tile 64x64 = 4x4 WMMA tiles; block 256x128.
template <bool BIAS, int OUTM, bool RESID>
__global__ __launch_bounds__(256)
void gemm_nt(const __bf16* __restrict__ A, const __bf16* __restrict__ B,
             const float* __restrict__ bias, const float* __restrict__ resid,
             void* __restrict__ Dp, int M, int N, int K, int ldd) {
  const int lane = threadIdx.x & 31;
  const int wave = threadIdx.x >> 5;
  const int m0 = blockIdx.y * 256 + (wave >> 1) * 64;
  const int n0 = blockIdx.x * 128 + (wave & 1) * 64;

  v8f zero = {0.f, 0.f, 0.f, 0.f, 0.f, 0.f, 0.f, 0.f};
  v8f acc[4][4];
#pragma unroll
  for (int i = 0; i < 4; ++i)
#pragma unroll
    for (int j = 0; j < 4; ++j) acc[i][j] = zero;

#pragma unroll 2
  for (int k0 = 0; k0 < K; k0 += 32) {
    // WGP-scope prefetch of next K slice (speculative; dropped at tensor end)
    __builtin_prefetch(A + (size_t)(m0 + (lane & 15)) * K + k0 + 32, 0, 3);
    __builtin_prefetch(B + (size_t)(n0 + (lane & 15)) * K + k0 + 32, 0, 3);

    v16bf af[4], bfr[4];
#pragma unroll
    for (int i = 0; i < 4; ++i) af[i]  = load_fragA(A, K, m0 + 16 * i, k0, lane);
#pragma unroll
    for (int j = 0; j < 4; ++j) bfr[j] = load_fragB(B, K, n0 + 16 * j, k0, lane);

#pragma unroll
    for (int i = 0; i < 4; ++i)
#pragma unroll
      for (int j = 0; j < 4; ++j)
        acc[i][j] = __builtin_amdgcn_wmma_f32_16x16x32_bf16(
            false, af[i], false, bfr[j], (short)0, acc[i][j], false, false);
  }

  // Epilogue. C/D layout: VGPR r of lane L -> row (8*(L>=16)+r), col (L&15).
  const int colt = lane & 15;
  const int rowh = (lane >> 4) << 3;
#pragma unroll
  for (int i = 0; i < 4; ++i) {
#pragma unroll
    for (int j = 0; j < 4; ++j) {
      const int n = n0 + 16 * j + colt;
      const float bval = BIAS ? bias[n] : 0.0f;
#pragma unroll
      for (int r = 0; r < 8; ++r) {
        const int m = m0 + 16 * i + rowh + r;
        float v = acc[i][j][r] + bval;
        if (OUTM == 0) {
          float* D = (float*)Dp;
          if (RESID) v += resid[(size_t)m * ldd + n];
          D[(size_t)m * ldd + n] = v;
        } else if (OUTM == 1) {
          ((__bf16*)Dp)[(size_t)m * ldd + n] = (__bf16)v;
        } else {
          ((__bf16*)Dp)[(size_t)n * ldd + m] = (__bf16)v;  // transposed store
        }
      }
    }
  }
}

// ------------------------------ row softmax --------------------------------
// One 256-thread block per row of S[4096]; 16 elems/thread held in registers.
__global__ __launch_bounds__(256)
void softmax_row_bf16(const float* __restrict__ S, __bf16* __restrict__ P, int n) {
  const int row  = blockIdx.x;
  const int tid  = threadIdx.x;
  const int lane = tid & 31;
  const int wave = tid >> 5;
  const float* s = S + (size_t)row * n;
  __bf16* p = P + (size_t)row * n;

  float vals[16];
  float mx = -3.402823466e38f;
#pragma unroll
  for (int e = 0; e < 16; ++e) {
    vals[e] = s[tid + e * 256];
    mx = fmaxf(mx, vals[e]);
  }
#pragma unroll
  for (int off = 16; off; off >>= 1) mx = fmaxf(mx, __shfl_xor(mx, off, 32));

  __shared__ float sred[8];
  if (lane == 0) sred[wave] = mx;
  __syncthreads();
  {
    float t = sred[lane & 7];
#pragma unroll
    for (int off = 4; off; off >>= 1) t = fmaxf(t, __shfl_xor(t, off, 32));
    mx = __shfl(t, 0, 32);
  }
  __syncthreads();  // sred reuse barrier

  float sum = 0.f;
#pragma unroll
  for (int e = 0; e < 16; ++e) {
    vals[e] = __expf(vals[e] - mx);
    sum += vals[e];
  }
#pragma unroll
  for (int off = 16; off; off >>= 1) sum += __shfl_xor(sum, off, 32);
  if (lane == 0) sred[wave] = sum;
  __syncthreads();
  {
    float t = sred[lane & 7];
#pragma unroll
    for (int off = 4; off; off >>= 1) t += __shfl_xor(t, off, 32);
    sum = __shfl(t, 0, 32);
  }
  const float inv = 1.0f / sum;
#pragma unroll
  for (int e = 0; e < 16; ++e) p[tid + e * 256] = (__bf16)(vals[e] * inv);
}

// ------------------------------- launcher ----------------------------------
extern "C" void kernel_launch(void* const* d_in, const int* in_sizes, int n_in,
                              void* d_out, int out_size, void* d_ws, size_t ws_size,
                              hipStream_t stream) {
  const float* x  = (const float*)d_in[0];
  const float* Wq = (const float*)d_in[1];
  const float* bq = (const float*)d_in[2];
  const float* Wk = (const float*)d_in[3];
  const float* bk = (const float*)d_in[4];
  const float* Wv = (const float*)d_in[5];
  const float* bv = (const float*)d_in[6];
  const float* Wo = (const float*)d_in[7];
  const float* bo = (const float*)d_in[8];
  float* out = (float*)d_out;

  char* ws = (char*)d_ws;
  const size_t SZ_X  = (size_t)NTOK * CDIM * 2;   // 8 MB bf16
  const size_t SZ_W  = (size_t)CDIM * CDIM * 2;   // 2 MB bf16
  __bf16* x_bf  = (__bf16*)(ws);
  __bf16* Wq_bf = (__bf16*)(ws + SZ_X);
  __bf16* Wk_bf = (__bf16*)(ws + SZ_X + 1 * SZ_W);
  __bf16* Wv_bf = (__bf16*)(ws + SZ_X + 2 * SZ_W);
  __bf16* Wo_bf = (__bf16*)(ws + SZ_X + 3 * SZ_W);
  __bf16* q_bf  = (__bf16*)(ws + SZ_X + 4 * SZ_W);
  __bf16* k_bf  = (__bf16*)(ws + 2 * SZ_X + 4 * SZ_W);
  __bf16* vT_bf = (__bf16*)(ws + 3 * SZ_X + 4 * SZ_W);
  __bf16* h_bf  = (__bf16*)(ws + 4 * SZ_X + 4 * SZ_W);
  float*  S     = (float*) (ws + 5 * SZ_X + 4 * SZ_W);                       // 64 MB
  __bf16* P     = (__bf16*)(ws + 5 * SZ_X + 4 * SZ_W +
                            (size_t)NTOK * NTOK * 4);                        // 32 MB

  // 1) fp32 -> bf16 conversions
  cvt_f32_bf16<<<4096, 256, 0, stream>>>(x, x_bf, NTOK * CDIM);
  cvt_f32_bf16<<<1024, 256, 0, stream>>>(Wq, Wq_bf, CDIM * CDIM);
  cvt_f32_bf16<<<1024, 256, 0, stream>>>(Wk, Wk_bf, CDIM * CDIM);
  cvt_f32_bf16<<<1024, 256, 0, stream>>>(Wv, Wv_bf, CDIM * CDIM);
  cvt_f32_bf16<<<1024, 256, 0, stream>>>(Wo, Wo_bf, CDIM * CDIM);

  // 2) projections: q,k (bf16), v stored transposed [C, N]
  dim3 gProj(CDIM / 128, NTOK / 256);  // (8, 16)
  gemm_nt<true, 1, false><<<gProj, 256, 0, stream>>>(x_bf, Wq_bf, bq, nullptr,
                                                     q_bf, NTOK, CDIM, CDIM, CDIM);
  gemm_nt<true, 1, false><<<gProj, 256, 0, stream>>>(x_bf, Wk_bf, bk, nullptr,
                                                     k_bf, NTOK, CDIM, CDIM, CDIM);
  gemm_nt<true, 2, false><<<gProj, 256, 0, stream>>>(x_bf, Wv_bf, bv, nullptr,
                                                     vT_bf, NTOK, CDIM, CDIM, NTOK);

  // 3) S = q k^T  (fp32, 4096x4096)
  dim3 gS(NTOK / 128, NTOK / 256);  // (32, 16)
  gemm_nt<false, 0, false><<<gS, 256, 0, stream>>>(q_bf, k_bf, nullptr, nullptr,
                                                   S, NTOK, NTOK, CDIM, NTOK);

  // 4) P = softmax(S) row-wise, bf16
  softmax_row_bf16<<<NTOK, 256, 0, stream>>>(S, P, NTOK);

  // 5) h = P v  ==  P[N,N] * (vT[C,N])^T  -> bf16
  gemm_nt<false, 1, false><<<gProj, 256, 0, stream>>>(P, vT_bf, nullptr, nullptr,
                                                      h_bf, NTOK, CDIM, NTOK, CDIM);

  // 6) out = x + h Wo^T + bo  (fp32)
  gemm_nt<true, 0, true><<<gProj, 256, 0, stream>>>(h_bf, Wo_bf, bo, x,
                                                    out, NTOK, CDIM, CDIM, CDIM);
}
